// ClassifierFramework_16527034155371
// MI455X (gfx1250) — matrix operand order repaired and checked by reading, not compiled, not run on hardware
//
#include <hip/hip_runtime.h>
#include <hip/hip_bf16.h>

// ---------------------------------------------------------------------------
// Problem constants (from reference setup_inputs)
// ---------------------------------------------------------------------------
#define BATCH   4096
#define NNODES  64
#define FIN     67
#define DFP     2048
#define NEDGE   512
#define NEDGE_T (NEDGE + NNODES)   // + self loops = 576
#define EPSBN   1e-5f

typedef __attribute__((ext_vector_type(16))) __bf16 v16bf;
typedef __attribute__((ext_vector_type(8)))  __bf16 v8bf;
typedef __attribute__((ext_vector_type(8)))  float  v8f;

// round-to-nearest-even float -> bf16 raw bits
__device__ __forceinline__ unsigned short f2bfu(float f) {
    unsigned u = __float_as_uint(f);
    return (unsigned short)((u + 0x7FFFu + ((u >> 16) & 1u)) >> 16);
}

__device__ __forceinline__ void store_val(float* p, float v) { *p = v; }
__device__ __forceinline__ void store_val(unsigned short* p, float v) { *p = f2bfu(v); }

// A fragment: element i -> K = kb + 16*half + i (matches B fragment).
// Two 16B loads + shuffle.
__device__ __forceinline__ v16bf load_frag_a(const unsigned short* Arow, int kb, int half) {
    const v8bf lo = *(const v8bf*)(Arow + kb + half * 16);
    const v8bf hi = *(const v8bf*)(Arow + kb + half * 16 + 8);
    return __builtin_shufflevector(lo, hi, 0,1,2,3,4,5,6,7,8,9,10,11,12,13,14,15);
}
// B fragment: element i -> K = kb + 16*half + i (one contiguous 32B run)
__device__ __forceinline__ v16bf load_frag_b(const unsigned short* Wrow, int kb, int half) {
    const v8bf lo = *(const v8bf*)(Wrow + kb + half * 16);
    const v8bf hi = *(const v8bf*)(Wrow + kb + half * 16 + 8);
    return __builtin_shufflevector(lo, hi, 0,1,2,3,4,5,6,7,8,9,10,11,12,13,14,15);
}

// ---------------------------------------------------------------------------
// f32 -> bf16 conversion pass with optional row padding (pad pre-zeroed by
// hipMemsetAsync). One row per block.
// ---------------------------------------------------------------------------
__global__ __launch_bounds__(128) void cvt_bf16_kernel(
    const float* __restrict__ src, unsigned short* __restrict__ dst,
    int srow, int drow)
{
    const size_t r = blockIdx.x;
    const float* s = src + r * (size_t)srow;
    unsigned short* d = dst + r * (size_t)drow;
    for (int c = threadIdx.x; c < srow; c += 128) d[c] = f2bfu(s[c]);
}

// ---------------------------------------------------------------------------
// WMMA GEMM:  C[M,N] = act( A[M,K]bf16 * W[N,K]bf16^T + bias )
// One 16x(16*NT) strip per wave; NT accumulators share one A fragment per
// K-step. The K-loop is software-pipelined AND unrolled by 2 with explicit
// ping-pong fragment buffers: buffer roles alternate structurally, so the
// register allocator needs no rotation moves, and each v_wmma group runs
// while the other buffer's b128 loads are in flight (partial s_wait_loadcnt).
// K is pre-padded to a multiple of 32 and all fragment addresses are 16B
// aligned. All branches are wave-uniform -> EXEC all-ones at every WMMA.
// ---------------------------------------------------------------------------
template <int NT, typename OUT>
__global__ __launch_bounds__(256) void wmma_gemm(
    const unsigned short* __restrict__ A, int lda,
    const unsigned short* __restrict__ W, int ldw,
    const float* __restrict__ bias,
    OUT* __restrict__ C, int ldc,
    int mtiles, int ntiles, int kmax, int relu)
{
    const int ngroups = (ntiles + NT - 1) / NT;
    const int wid = blockIdx.x * 8 + (threadIdx.x >> 5);   // wave id (uniform)
    const int tg  = wid % ngroups;
    const int tm  = wid / ngroups;
    if (tm >= mtiles) return;                              // wave-uniform exit

    const int lane = threadIdx.x & 31;
    const int half = lane >> 4;      // 0: lanes 0-15, 1: lanes 16-31
    const int l15  = lane & 15;

    const unsigned short* Arow = A + (size_t)(tm * 16 + l15) * (size_t)lda;
    const unsigned short* Wr[NT];
    #pragma unroll
    for (int j = 0; j < NT; ++j) {
        int tn = tg * NT + j;
        if (tn > ntiles - 1) tn = ntiles - 1;              // clamp (uniform)
        Wr[j] = W + (size_t)(tn * 16 + l15) * (size_t)ldw;
    }

    v8f c[NT];
    #pragma unroll
    for (int j = 0; j < NT; ++j) c[j] = (v8f){};

    const int ksteps = kmax >> 5;

    v16bf a0, a1, b0[NT], b1[NT];

    // prologue: buffer 0 holds K-step 0
    a0 = load_frag_a(Arow, 0, half);
    #pragma unroll
    for (int j = 0; j < NT; ++j) b0[j] = load_frag_b(Wr[j], 0, half);

    // main loop, 2 K-steps per iteration, ping-pong buffers (no reg copies)
    int ks = 0;
    for (; ks + 2 < ksteps; ks += 2) {
        const int k1 = (ks + 1) << 5;
        const int k2 = (ks + 2) << 5;
        a1 = load_frag_a(Arow, k1, half);
        #pragma unroll
        for (int j = 0; j < NT; ++j) b1[j] = load_frag_b(Wr[j], k1, half);
        #pragma unroll
        for (int j = 0; j < NT; ++j)
            c[j] = __builtin_amdgcn_wmma_f32_16x16x32_bf16(
                       false, a0, false, b0[j], (short)0, c[j], false, false);
        a0 = load_frag_a(Arow, k2, half);
        #pragma unroll
        for (int j = 0; j < NT; ++j) b0[j] = load_frag_b(Wr[j], k2, half);
        #pragma unroll
        for (int j = 0; j < NT; ++j)
            c[j] = __builtin_amdgcn_wmma_f32_16x16x32_bf16(
                       false, a1, false, b1[j], (short)0, c[j], false, false);
    }
    // tail: 1 or 2 steps remain, buffer 0 holds step ks
    if (ks + 1 < ksteps) {
        const int k1 = (ks + 1) << 5;
        a1 = load_frag_a(Arow, k1, half);
        #pragma unroll
        for (int j = 0; j < NT; ++j) b1[j] = load_frag_b(Wr[j], k1, half);
        #pragma unroll
        for (int j = 0; j < NT; ++j)
            c[j] = __builtin_amdgcn_wmma_f32_16x16x32_bf16(
                       false, a0, false, b0[j], (short)0, c[j], false, false);
        #pragma unroll
        for (int j = 0; j < NT; ++j)
            c[j] = __builtin_amdgcn_wmma_f32_16x16x32_bf16(
                       false, a1, false, b1[j], (short)0, c[j], false, false);
    } else {
        #pragma unroll
        for (int j = 0; j < NT; ++j)
            c[j] = __builtin_amdgcn_wmma_f32_16x16x32_bf16(
                       false, a0, false, b0[j], (short)0, c[j], false, false);
    }

    #pragma unroll
    for (int j = 0; j < NT; ++j) {
        const int tn = tg * NT + j;
        if (tn >= ntiles) break;                           // uniform guard
        const int   ncol = tn * 16 + l15;
        const float bi   = bias ? bias[ncol] : 0.0f;
        #pragma unroll
        for (int r = 0; r < 8; ++r) {
            const int m = tm * 16 + half * 8 + r;  // C/D: VGPR r -> M = 8*half + r
            float v = c[j][r] + bi;
            if (relu) v = fmaxf(v, 0.0f);
            store_val(&C[(size_t)m * (size_t)ldc + ncol], v);
        }
    }
}

// ---------------------------------------------------------------------------
// Edge normalization: deg of (A + I) via dst counts, norm[e]=dinv[src]*dinv[dst]
// ---------------------------------------------------------------------------
__global__ __launch_bounds__(256) void norm_kernel(
    const int* __restrict__ el, float* __restrict__ nrm)
{
    __shared__ float deg[NNODES];
    __shared__ float dinv[NNODES];
    const int t = threadIdx.x;
    if (t < NNODES) deg[t] = 0.0f;
    __syncthreads();
    for (int e = t; e < NEDGE_T; e += 256) {
        int d = (e < NEDGE) ? el[NEDGE + e] : (e - NEDGE);
        atomicAdd(&deg[d], 1.0f);
    }
    __syncthreads();
    if (t < NNODES) dinv[t] = (deg[t] > 0.0f) ? rsqrtf(deg[t]) : 0.0f;
    __syncthreads();
    for (int e = t; e < NEDGE_T; e += 256) {
        int s = (e < NEDGE) ? el[e] : (e - NEDGE);
        int d = (e < NEDGE) ? el[NEDGE + e] : (e - NEDGE);
        nrm[e] = dinv[s] * dinv[d];
    }
}

// ---------------------------------------------------------------------------
// Per-batch scatter-add aggregation, in-place on G (+ bias). One batch per
// block, one feature per thread; whole 64xF batch tile staged in LDS so the
// serial edge loop is conflict-free (threads own disjoint feature columns).
// ---------------------------------------------------------------------------
__global__ void agg_kernel(const int* __restrict__ el,
                           const float* __restrict__ nrm,
                           const float* __restrict__ bias,
                           float* G, int F)
{
    extern __shared__ float smem[];
    float* h   = smem;                 // [64*F]
    float* acc = smem + NNODES * F;    // [64*F]
    int*   es  = (int*)(smem + 2 * NNODES * F);
    int*   ed  = es + NEDGE_T;
    float* nw  = (float*)(ed + NEDGE_T);

    const int f = threadIdx.x;            // blockDim.x == F
    const int b = blockIdx.x;
    float* Gb = G + (size_t)b * NNODES * F;

    for (int i = f; i < NNODES * F; i += F) { h[i] = Gb[i]; acc[i] = 0.0f; }
    for (int e = f; e < NEDGE_T; e += F) {
        es[e] = (e < NEDGE) ? el[e] : (e - NEDGE);
        ed[e] = (e < NEDGE) ? el[NEDGE + e] : (e - NEDGE);
        nw[e] = nrm[e];
    }
    __syncthreads();
    for (int e = 0; e < NEDGE_T; ++e)
        acc[ed[e] * F + f] += nw[e] * h[es[e] * F + f];
    __syncthreads();
    const float bi = bias[f];
    for (int n = 0; n < NNODES; ++n)
        Gb[n * F + f] = acc[n * F + f] + bi;
}

// ---------------------------------------------------------------------------
// BatchNorm stats per node n: mean/var over (batch, feature) = 4096*F values.
// ---------------------------------------------------------------------------
__global__ __launch_bounds__(256) void stats_kernel(
    const float* __restrict__ G, int F, int fshift, float* __restrict__ meanvar)
{
    const int n = blockIdx.x;
    const int t = threadIdx.x;
    float s = 0.0f, sq = 0.0f;
    const int total = BATCH * F;
    for (int i = t; i < total; i += 256) {
        int b = i >> fshift;
        int f = i & (F - 1);
        float v = G[((size_t)b * NNODES + n) * F + f];
        s += v; sq += v * v;
    }
    __shared__ float rs[256], rq[256];
    rs[t] = s; rq[t] = sq;
    __syncthreads();
    for (int o = 128; o > 0; o >>= 1) {
        if (t < o) { rs[t] += rs[t + o]; rq[t] += rq[t + o]; }
        __syncthreads();
    }
    if (t == 0) {
        float inv = 1.0f / (float)total;
        float m = rs[0] * inv;
        meanvar[n]          = m;
        meanvar[NNODES + n] = rq[0] * inv - m * m;   // population var
    }
}

// ---------------------------------------------------------------------------
// BN (per-node) + ReLU, elementwise; OUT = float (pre-pool) or bf16 (GEMM feed)
// ---------------------------------------------------------------------------
template <typename OUT>
__global__ __launch_bounds__(256) void bn_relu_kernel(
    const float* __restrict__ X, OUT* __restrict__ Y,
    const float* __restrict__ meanvar,
    const float* __restrict__ gamma, const float* __restrict__ beta,
    int fshift)
{
    const size_t i = (size_t)blockIdx.x * 256 + threadIdx.x;
    const int n = (int)((i >> fshift) & (NNODES - 1));
    const float m = meanvar[n];
    const float v = meanvar[NNODES + n];
    float y = (X[i] - m) * rsqrtf(v + EPSBN) * gamma[n] + beta[n];
    store_val(&Y[i], fmaxf(y, 0.0f));
}

// global max pool over nodes: pooled[b][f] = max_n A[b][n][f], F = 32
__global__ __launch_bounds__(256) void pool_kernel(
    const float* __restrict__ A, float* __restrict__ pooled)
{
    const int i = blockIdx.x * 256 + threadIdx.x;   // b*32 + f
    const int b = i >> 5, f = i & 31;
    float mx = -3.4e38f;
    for (int n = 0; n < NNODES; ++n)
        mx = fmaxf(mx, A[((size_t)b * NNODES + n) * 32 + f]);
    pooled[i] = mx;
}

// head: out[b,o] = [pooled | hmlp] . Wfc[o] + bfc[o]   (96 -> 2)
__global__ __launch_bounds__(256) void head_kernel(
    const float* __restrict__ pooled, const float* __restrict__ hm,
    const float* __restrict__ Wfc, const float* __restrict__ bfc,
    float* __restrict__ out)
{
    const int b = blockIdx.x * 256 + threadIdx.x;
    if (b >= BATCH) return;
    #pragma unroll
    for (int o = 0; o < 2; ++o) {
        float s = bfc[o];
        for (int f = 0; f < 32; ++f) s += pooled[b * 32 + f] * Wfc[o * 96 + f];
        for (int j = 0; j < 64; ++j) s += hm[b * 64 + j] * Wfc[o * 96 + 32 + j];
        out[b * 2 + o] = s;
    }
}

// ---------------------------------------------------------------------------
// Orchestration
// ---------------------------------------------------------------------------
extern "C" void kernel_launch(void* const* d_in, const int* in_sizes, int n_in,
                              void* d_out, int out_size, void* d_ws, size_t ws_size,
                              hipStream_t stream) {
    const float* xfp  = (const float*)d_in[0];   // [4096,2048]
    const float* xnd  = (const float*)d_in[1];   // [4096,64,67]
    const int*   el   = (const int*)  d_in[2];   // [2,512]
    const float* W1   = (const float*)d_in[3];   // [64,67]
    const float* b1   = (const float*)d_in[4];
    const float* g1   = (const float*)d_in[5];
    const float* be1  = (const float*)d_in[6];
    const float* W2   = (const float*)d_in[7];   // [32,64]
    const float* b2   = (const float*)d_in[8];
    const float* g2   = (const float*)d_in[9];
    const float* be2  = (const float*)d_in[10];
    const float* Wl1  = (const float*)d_in[11];  // [400,2048]
    const float* bl1  = (const float*)d_in[12];
    const float* Wl2  = (const float*)d_in[13];  // [64,400]
    const float* bl2  = (const float*)d_in[14];
    const float* Wfc  = (const float*)d_in[15];  // [2,96]
    const float* bfc  = (const float*)d_in[16];
    float* out = (float*)d_out;                  // [4096,2]

    // ---- workspace layout (byte offsets, 256B aligned regions) -------------
    char* wsb = (char*)d_ws;
    float*          NRM  = (float*)(wsb);                       // 576 f32
    float*          ST1  = (float*)(wsb + 4096);                // mean+var [128]
    float*          ST2  = (float*)(wsb + 8192);
    unsigned short* XFPB = (unsigned short*)(wsb + 16384);      // [4096,2048] bf16
    size_t off = 16384 + (size_t)BATCH * DFP * 2;
    unsigned short* WL1B = (unsigned short*)(wsb + off); off += (size_t)400 * DFP * 2;
    unsigned short* WL2B = (unsigned short*)(wsb + off); off += (size_t)64 * 416 * 2;
    unsigned short* W1B  = (unsigned short*)(wsb + off); off += (size_t)64 * 96 * 2;
    unsigned short* W2B  = (unsigned short*)(wsb + off); off += (size_t)32 * 64 * 2;
    unsigned short* XNDB = (unsigned short*)(wsb + off); off += (size_t)BATCH * NNODES * 96 * 2;
    unsigned short* H1   = (unsigned short*)(wsb + off); off += (size_t)BATCH * 416 * 2;
    float*          HM2  = (float*)(wsb + off);          off += (size_t)BATCH * 64 * 4;
    float*          G1   = (float*)(wsb + off);          off += (size_t)BATCH * 64 * 64 * 4;
    float*          A2   = G1 + (size_t)BATCH * 64 * 32;        // upper half of G1 region
    unsigned short* A1N  = (unsigned short*)(wsb + off); off += (size_t)BATCH * 64 * 64 * 2;
    float*          POOL = (float*)(wsb + off);          off += (size_t)BATCH * 32 * 4;
    // total ~174 MB

    (void)in_sizes; (void)n_in; (void)out_size; (void)ws_size;

    // zero K-pad regions consumed by GEMMs (graph-capturable memset nodes)
    hipMemsetAsync(XNDB, 0, (size_t)BATCH * NNODES * 96 * 2, stream);
    hipMemsetAsync(H1,   0, (size_t)BATCH * 416 * 2, stream);
    hipMemsetAsync(WL2B, 0, (size_t)64 * 416 * 2, stream);
    hipMemsetAsync(W1B,  0, (size_t)64 * 96 * 2, stream);

    // one-time f32 -> bf16 conversions (bandwidth-bound pre-pass)
    cvt_bf16_kernel<<<BATCH, 128, 0, stream>>>(xfp, XFPB, DFP, DFP);
    cvt_bf16_kernel<<<400, 128, 0, stream>>>(Wl1, WL1B, DFP, DFP);
    cvt_bf16_kernel<<<64, 128, 0, stream>>>(Wl2, WL2B, 400, 416);
    cvt_bf16_kernel<<<64, 128, 0, stream>>>(W1, W1B, FIN, 96);
    cvt_bf16_kernel<<<32, 128, 0, stream>>>(W2, W2B, 64, 64);
    cvt_bf16_kernel<<<BATCH * NNODES, 128, 0, stream>>>(xnd, XNDB, FIN, 96);

    // edge norms for (A + I)
    norm_kernel<<<1, 256, 0, stream>>>(el, NRM);

    // ---- fingerprint MLP: relu(Xfp @ Wl1^T + bl1) -> H1 bf16 [4096,400(416)]
    {
        int mt = BATCH / 16, nt = 25, ng = (nt + 3) / 4;
        wmma_gemm<4, unsigned short><<<(mt * ng + 7) / 8, 256, 0, stream>>>(
            XFPB, DFP, WL1B, DFP, bl1, H1, 416, mt, nt, DFP, 1);
    }
    // ---- relu(H1 @ Wl2^T + bl2) -> HM2 f32 [4096,64]   (K padded to 416)
    {
        int mt = BATCH / 16, nt = 4;
        wmma_gemm<4, float><<<(mt + 7) / 8, 256, 0, stream>>>(
            H1, 416, WL2B, 416, bl2, HM2, 64, mt, nt, 416, 1);
    }
    // ---- GCN layer 1 matmul: [B*64,67(96)] @ W1^T -> G1 f32 [B*64,64]
    {
        int mt = BATCH * NNODES / 16, nt = 4;
        wmma_gemm<4, float><<<(mt + 7) / 8, 256, 0, stream>>>(
            XNDB, 96, W1B, 96, nullptr, G1, 64, mt, nt, 96, 0);
    }
    // scatter-add over edges (+b1), in place on G1
    {
        size_t smem = (size_t)(2 * NNODES * 64) * 4 + NEDGE_T * 12;
        agg_kernel<<<BATCH, 64, smem, stream>>>(el, NRM, b1, G1, 64);
    }
    stats_kernel<<<NNODES, 256, 0, stream>>>(G1, 64, 6, ST1);
    bn_relu_kernel<unsigned short><<<(BATCH * NNODES * 64) / 256, 256, 0, stream>>>(
        G1, A1N, ST1, g1, be1, 6);

    // ---- GCN layer 2 matmul: [B*64,64] @ W2^T -> G2(=G1) f32 [B*64,32]
    {
        int mt = BATCH * NNODES / 16, nt = 2;
        wmma_gemm<2, float><<<(mt + 7) / 8, 256, 0, stream>>>(
            A1N, 64, W2B, 64, nullptr, G1, 32, mt, nt, 64, 0);
    }
    {
        size_t smem = (size_t)(2 * NNODES * 32) * 4 + NEDGE_T * 12;
        agg_kernel<<<BATCH, 32, smem, stream>>>(el, NRM, b2, G1, 32);
    }
    stats_kernel<<<NNODES, 256, 0, stream>>>(G1, 32, 5, ST2);
    bn_relu_kernel<float><<<(BATCH * NNODES * 32) / 256, 256, 0, stream>>>(
        G1, A2, ST2, g2, be2, 5);

    // max pool over nodes -> POOL [4096,32]
    pool_kernel<<<(BATCH * 32) / 256, 256, 0, stream>>>(A2, POOL);

    // final linear head
    head_kernel<<<(BATCH + 255) / 256, 256, 0, stream>>>(POOL, HM2, Wfc, bfc, out);
}
